// FWMRNN_69020124446844
// MI455X (gfx1250) — compile-verified, matrix operand order checked by reading, not run on hardware
//
#include <hip/hip_runtime.h>
#include <hip/hip_bf16.h>
#include <math.h>

// FWM-RNN for MI455X (gfx1250): bf16 WMMA GEMMs (double-buffered, spill-free tiling),
// split-K recurrent GEMM for latency-hiding parallelism, VALU FWM.

typedef __attribute__((ext_vector_type(16))) __bf16 v16bf;
typedef __attribute__((ext_vector_type(8)))  float  v8f;

#define L_SEQ 256
#define B_SZ  16
#define I_SZ  2048
#define H_SZ  2048
#define G_SZ  8192   /* 4*H */
#define SR_SZ 1024   /* S*R */
#define T_SZ  32
#define KSPLIT 8     /* K-splits for recurrent GEMM */

#define WMMA_BF16(A, Bv, C) \
  __builtin_amdgcn_wmma_f32_16x16x32_bf16(false, (A), false, (Bv), (short)0, (C), false, false)

__device__ __forceinline__ __bf16 f2bf(float f) {
  union { float f; unsigned u; } v; v.f = f;
  unsigned r = v.u + 0x7FFFu + ((v.u >> 16) & 1u);   // round-to-nearest-even
  unsigned short h = (unsigned short)(r >> 16);
  return __builtin_bit_cast(__bf16, h);
}
__device__ __forceinline__ float sigm(float x) { return 1.0f / (1.0f + __expf(-x)); }

// ---------------- conversion / init ----------------
__global__ void k_f32_to_bf16(const float* __restrict__ src, __bf16* __restrict__ dst, int n) {
  for (int i = blockIdx.x * blockDim.x + threadIdx.x; i < n; i += gridDim.x * blockDim.x)
    dst[i] = f2bf(src[i]);
}
__global__ void k_copy_f32(const float* __restrict__ src, float* __restrict__ dst, int n) {
  for (int i = blockIdx.x * blockDim.x + threadIdx.x; i < n; i += gridDim.x * blockDim.x)
    dst[i] = src[i];
}

// ---------------- Gx = inputs @ W_ih^T + (b_ih+b_hh) ----------------
// M = 4096, N = 8192, K = 2048. One wave = 32x32 tile (4 accums, ~110 VGPRs live:
// fits without spills), register double-buffered fragments, peeled tail, WGP prefetch.
__global__ void __launch_bounds__(128)
k_gx_gemm(const __bf16* __restrict__ Xbf, const __bf16* __restrict__ W,
          const float* __restrict__ bih, const float* __restrict__ bhh,
          float* __restrict__ Gx) {
  const int wave = blockIdx.x * (blockDim.x >> 5) + (threadIdx.x >> 5);
  const int lane = threadIdx.x & 31;
  const int mt = wave >> 8;          // 128 M-groups of 32
  const int nq = wave & 255;         // 256 N-groups of 32
  const int m0 = mt * 32;
  const int n0 = nq * 32;
  const int r16  = lane & 15;
  const int koff = (lane >> 4) * 16;

  const __bf16* A0 = Xbf + (size_t)(m0 + r16) * I_SZ + koff;
  const __bf16* A1 = A0 + (size_t)16 * I_SZ;
  const __bf16* B0 = W + (size_t)(n0 + r16) * I_SZ + koff;
  const __bf16* B1 = B0 + (size_t)16 * I_SZ;

  v8f c00 = {}, c01 = {}, c10 = {}, c11 = {};

  v16bf aA0 = *(const v16bf*)(A0), aA1 = *(const v16bf*)(A1);
  v16bf bA0 = *(const v16bf*)(B0), bA1 = *(const v16bf*)(B1);

  int k = 0;
  for (; k < I_SZ - 64; k += 64) {
    const int kb = k + 32;
    v16bf aB0 = *(const v16bf*)(A0 + kb), aB1 = *(const v16bf*)(A1 + kb);
    v16bf bB0 = *(const v16bf*)(B0 + kb), bB1 = *(const v16bf*)(B1 + kb);
    __builtin_prefetch(B0 + k + 256, 0, 3);       // WGP-scope stream-ahead
    __builtin_prefetch(B1 + k + 256, 0, 3);
    c00 = WMMA_BF16(aA0, bA0, c00); c01 = WMMA_BF16(aA0, bA1, c01);
    c10 = WMMA_BF16(aA1, bA0, c10); c11 = WMMA_BF16(aA1, bA1, c11);
    const int ka = k + 64;
    aA0 = *(const v16bf*)(A0 + ka); aA1 = *(const v16bf*)(A1 + ka);
    bA0 = *(const v16bf*)(B0 + ka); bA1 = *(const v16bf*)(B1 + ka);
    c00 = WMMA_BF16(aB0, bB0, c00); c01 = WMMA_BF16(aB0, bB1, c01);
    c10 = WMMA_BF16(aB1, bB0, c10); c11 = WMMA_BF16(aB1, bB1, c11);
  }
  { // peeled final pair (k = I_SZ-64)
    const int kb = k + 32;
    v16bf aB0 = *(const v16bf*)(A0 + kb), aB1 = *(const v16bf*)(A1 + kb);
    v16bf bB0 = *(const v16bf*)(B0 + kb), bB1 = *(const v16bf*)(B1 + kb);
    c00 = WMMA_BF16(aA0, bA0, c00); c01 = WMMA_BF16(aA0, bA1, c01);
    c10 = WMMA_BF16(aA1, bA0, c10); c11 = WMMA_BF16(aA1, bA1, c11);
    c00 = WMMA_BF16(aB0, bB0, c00); c01 = WMMA_BF16(aB0, bB1, c01);
    c10 = WMMA_BF16(aB1, bB0, c10); c11 = WMMA_BF16(aB1, bB1, c11);
  }

  // C/D layout: VGPR j, lanes 0-15 -> M=j, lanes 16-31 -> M=j+8; N = lane&15
  const int mb = (lane >> 4) << 3;
  const int nb = lane & 15;
#pragma unroll
  for (int j = 0; j < 8; ++j) {
    int m, n;
    m = m0 + mb + j;
    n = n0 +  0 + nb; Gx[(size_t)m * G_SZ + n] = c00[j] + bih[n] + bhh[n];
    n = n0 + 16 + nb; Gx[(size_t)m * G_SZ + n] = c01[j] + bih[n] + bhh[n];
    m = m0 + 16 + mb + j;
    n = n0 +  0 + nb; Gx[(size_t)m * G_SZ + n] = c10[j] + bih[n] + bhh[n];
    n = n0 + 16 + nb; Gx[(size_t)m * G_SZ + n] = c11[j] + bih[n] + bhh[n];
  }
}

// ---------------- recurrent GEMM, split-K partials ----------------
// 1024 waves: 128 fused gate-column tiles x 8 K-splits (K=256 each).
__global__ void __launch_bounds__(128)
k_lstm_gemm(const __bf16* __restrict__ Hbf_in, const __bf16* __restrict__ Whh,
            float* __restrict__ Gpart) {
  const int wave = blockIdx.x * (blockDim.x >> 5) + (threadIdx.x >> 5);  // 0..1023
  const int lane = threadIdx.x & 31;
  const int ks = wave >> 7;          // K-split 0..7
  const int ct = wave & 127;         // column tile 0..127
  const int n0 = ct * 16;
  const int kbase = ks * (H_SZ / KSPLIT);   // 256-wide K slice
  const int r16  = lane & 15;
  const int koff = (lane >> 4) * 16;

  const __bf16* A  = Hbf_in + (size_t)r16 * H_SZ + kbase + koff;
  const __bf16* Bi = Whh + (size_t)(0 * H_SZ + n0 + r16) * H_SZ + kbase + koff;
  const __bf16* Bf = Whh + (size_t)(1 * H_SZ + n0 + r16) * H_SZ + kbase + koff;
  const __bf16* Bg = Whh + (size_t)(2 * H_SZ + n0 + r16) * H_SZ + kbase + koff;
  const __bf16* Bo = Whh + (size_t)(3 * H_SZ + n0 + r16) * H_SZ + kbase + koff;

  v8f ci = {}, cf = {}, cg = {}, co = {};

  v16bf aA  = *(const v16bf*)(A);
  v16bf biA = *(const v16bf*)(Bi), bfA = *(const v16bf*)(Bf);
  v16bf bgA = *(const v16bf*)(Bg), boA = *(const v16bf*)(Bo);

  const int KLEN = H_SZ / KSPLIT;    // 256
  int k = 0;
  for (; k < KLEN - 64; k += 64) {
    const int kb = k + 32;
    v16bf aB  = *(const v16bf*)(A + kb);
    v16bf biB = *(const v16bf*)(Bi + kb), bfB = *(const v16bf*)(Bf + kb);
    v16bf bgB = *(const v16bf*)(Bg + kb), boB = *(const v16bf*)(Bo + kb);
    __builtin_prefetch(Bi + k + 128, 0, 3);
    __builtin_prefetch(Bf + k + 128, 0, 3);
    __builtin_prefetch(Bg + k + 128, 0, 3);
    __builtin_prefetch(Bo + k + 128, 0, 3);
    ci = WMMA_BF16(aA, biA, ci); cf = WMMA_BF16(aA, bfA, cf);
    cg = WMMA_BF16(aA, bgA, cg); co = WMMA_BF16(aA, boA, co);
    const int ka = k + 64;
    aA  = *(const v16bf*)(A + ka);
    biA = *(const v16bf*)(Bi + ka); bfA = *(const v16bf*)(Bf + ka);
    bgA = *(const v16bf*)(Bg + ka); boA = *(const v16bf*)(Bo + ka);
    ci = WMMA_BF16(aB, biB, ci); cf = WMMA_BF16(aB, bfB, cf);
    cg = WMMA_BF16(aB, bgB, cg); co = WMMA_BF16(aB, boB, co);
  }
  { // peeled final pair (k = KLEN-64)
    const int kb = k + 32;
    v16bf aB  = *(const v16bf*)(A + kb);
    v16bf biB = *(const v16bf*)(Bi + kb), bfB = *(const v16bf*)(Bf + kb);
    v16bf bgB = *(const v16bf*)(Bg + kb), boB = *(const v16bf*)(Bo + kb);
    ci = WMMA_BF16(aA, biA, ci); cf = WMMA_BF16(aA, bfA, cf);
    cg = WMMA_BF16(aA, bgA, cg); co = WMMA_BF16(aA, boA, co);
    ci = WMMA_BF16(aB, biB, ci); cf = WMMA_BF16(aB, bfB, cf);
    cg = WMMA_BF16(aB, bgB, cg); co = WMMA_BF16(aB, boB, co);
  }

  const int mb = (lane >> 4) << 3;
  const int nb = lane & 15;
#pragma unroll
  for (int j = 0; j < 8; ++j) {
    const int m = mb + j;            // batch 0..15
    const int n = n0 + nb;
    float* gp = Gpart + (size_t)(ks * B_SZ + m) * G_SZ;
    gp[0 * H_SZ + n] = ci[j];
    gp[1 * H_SZ + n] = cf[j];
    gp[2 * H_SZ + n] = cg[j];
    gp[3 * H_SZ + n] = co[j];
  }
}

// ---------------- reduce partials + LSTM gates ----------------
__global__ void k_lstm_gate(const float* __restrict__ Gx_t, const float* __restrict__ Gpart,
                            float* __restrict__ Cst, __bf16* __restrict__ Hbf_out,
                            float* __restrict__ Xout_t) {
  const int idx = blockIdx.x * blockDim.x + threadIdx.x;   // 0..32767
  if (idx >= B_SZ * H_SZ) return;
  const int m = idx >> 11;
  const int n = idx & (H_SZ - 1);
  float gi = Gx_t[(size_t)m * G_SZ + 0 * H_SZ + n];
  float gf = Gx_t[(size_t)m * G_SZ + 1 * H_SZ + n];
  float gg = Gx_t[(size_t)m * G_SZ + 2 * H_SZ + n];
  float go = Gx_t[(size_t)m * G_SZ + 3 * H_SZ + n];
#pragma unroll
  for (int s = 0; s < KSPLIT; ++s) {
    const float* gp = Gpart + (size_t)(s * B_SZ + m) * G_SZ;
    gi += gp[0 * H_SZ + n];
    gf += gp[1 * H_SZ + n];
    gg += gp[2 * H_SZ + n];
    go += gp[3 * H_SZ + n];
  }
  const float cold = Cst[idx];
  const float cnew = sigm(gf) * cold + sigm(gi) * tanhf(gg);
  const float h    = sigm(go) * tanhf(cnew);
  Cst[idx]     = cnew;
  Hbf_out[idx] = f2bf(h);
  Xout_t[idx]  = h;
}

// ---------------- fast-weight-memory step: one workgroup per batch ----------------
__global__ void k_fwm_step(const float* __restrict__ Xt,
                           const float* __restrict__ Www, const float* __restrict__ Wwb,
                           const float* __restrict__ Wrw, const float* __restrict__ Wrb,
                           float* __restrict__ Fm, float* __restrict__ qs_t) {
  const int b = blockIdx.x;
  const int tid = threadIdx.x;          // 0..255 (8 waves)
  const int wv = tid >> 5, lane = tid & 31;
  const float* x = Xt + b * H_SZ;
  float* F = Fm + (size_t)b * SR_SZ * T_SZ;

  __shared__ float sw[97];
  __shared__ float ss[32], sr_[32], st[32];
  __shared__ float sbeta, sscale;
  __shared__ float ssr[SR_SZ];
  __shared__ float sv[32];
  __shared__ float srd[256];
  __shared__ float srv[128];
  __shared__ float sq[32], sqn[32];

  // w = x @ Ww_w^T + Ww_b
  for (int o = wv; o < 97; o += 8) {
    float sum = 0.f;
    for (int k = lane; k < H_SZ; k += 32) sum += x[k] * Www[o * H_SZ + k];
    for (int off = 16; off; off >>= 1) sum += __shfl_xor(sum, off, 32);
    if (lane == 0) sw[o] = sum + Wwb[o];
  }
  __syncthreads();
  if (tid < 32) { ss[tid] = tanhf(sw[tid]); sr_[tid] = tanhf(sw[32 + tid]); st[tid] = tanhf(sw[64 + tid]); }
  if (tid == 0) sbeta = sigm(sw[96] + 1.0f);
  __syncthreads();
  for (int k = tid; k < SR_SZ; k += 256) ssr[k] = ss[k >> 5] * sr_[k & 31];
  __syncthreads();
  // v = sr . Fm
  for (int tt = wv; tt < 32; tt += 8) {
    float sum = 0.f;
    for (int k = lane; k < SR_SZ; k += 32) sum += ssr[k] * F[k * T_SZ + tt];
    for (int off = 16; off; off >>= 1) sum += __shfl_xor(sum, off, 32);
    if (lane == 0) sv[tt] = sum;
  }
  __syncthreads();
  // Fm += sr (x) beta*(t - v), accumulate ||Fm||^2
  float loc = 0.f;
  for (int idx = tid; idx < SR_SZ * T_SZ; idx += 256) {
    const int k = idx >> 5, tt = idx & 31;
    const float f = F[idx] + ssr[k] * sbeta * (st[tt] - sv[tt]);
    F[idx] = f;
    loc += f * f;
  }
  srd[tid] = loc; __syncthreads();
  for (int s2 = 128; s2; s2 >>= 1) { if (tid < s2) srd[tid] += srd[tid + s2]; __syncthreads(); }
  if (tid == 0) { const float nn = sqrtf(srd[0]); sscale = 1.0f / (fmaxf(nn - 1.0f, 0.0f) + 1.0f); }
  __syncthreads();
  for (int idx = tid; idx < SR_SZ * T_SZ; idx += 256) F[idx] *= sscale;
  __syncthreads();
  // rv = tanh(x @ Wr_w^T + Wr_b)
  for (int o = wv; o < 128; o += 8) {
    float sum = 0.f;
    for (int k = lane; k < H_SZ; k += 32) sum += x[k] * Wrw[o * H_SZ + k];
    for (int off = 16; off; off >>= 1) sum += __shfl_xor(sum, off, 32);
    if (lane == 0) srv[o] = tanhf(sum + Wrb[o]);
  }
  __syncthreads();
  if (tid < 32) sq[tid] = srv[tid];
  __syncthreads();
  // 3 read-out hops, each followed by layernorm over 32
  for (int j = 0; j < 3; ++j) {
    for (int tt = wv; tt < 32; tt += 8) {
      float sum = 0.f;
      for (int k = lane; k < SR_SZ; k += 32)
        sum += sq[k >> 5] * srv[32 + j * 32 + (k & 31)] * F[k * T_SZ + tt];
      for (int off = 16; off; off >>= 1) sum += __shfl_xor(sum, off, 32);
      if (lane == 0) sqn[tt] = sum;
    }
    __syncthreads();
    if (tid < 32) {                              // whole wave0 active -> shuffles OK
      const float v = sqn[tid];
      float sum = v;
      for (int off = 16; off; off >>= 1) sum += __shfl_xor(sum, off, 32);
      const float mean = sum * (1.0f / 32.0f);
      const float d = v - mean;
      float vv = d * d;
      for (int off = 16; off; off >>= 1) vv += __shfl_xor(vv, off, 32);
      sq[tid] = d * rsqrtf(vv * (1.0f / 32.0f) + 1e-5f);
    }
    __syncthreads();
  }
  if (tid < 32) qs_t[b * T_SZ + tid] = sq[tid];
}

// ---------------- out += qs @ Wlin^T + Wlin_b ----------------
__global__ void k_final(const float* __restrict__ qs, const float* __restrict__ Wlin,
                        const float* __restrict__ blin, float* __restrict__ out, int n) {
  for (int i = blockIdx.x * blockDim.x + threadIdx.x; i < n; i += gridDim.x * blockDim.x) {
    const int r = i >> 11;
    const int h = i & 2047;
    const float* q = qs + r * T_SZ;
    const float* w = Wlin + h * T_SZ;
    float acc = blin[h];
#pragma unroll 8
    for (int t2 = 0; t2 < T_SZ; ++t2) acc += q[t2] * w[t2];
    out[i] += acc;
  }
}

extern "C" void kernel_launch(void* const* d_in, const int* in_sizes, int n_in,
                              void* d_out, int out_size, void* d_ws, size_t ws_size,
                              hipStream_t stream) {
  (void)in_sizes; (void)n_in; (void)out_size; (void)ws_size;
  const float* inputs = (const float*)d_in[0];
  const float* h0     = (const float*)d_in[1];
  const float* c0     = (const float*)d_in[2];
  const float* F0     = (const float*)d_in[3];
  const float* W_ih   = (const float*)d_in[4];
  const float* W_hh   = (const float*)d_in[5];
  const float* b_ih   = (const float*)d_in[6];
  const float* b_hh   = (const float*)d_in[7];
  const float* Ww_w   = (const float*)d_in[8];
  const float* Ww_b   = (const float*)d_in[9];
  const float* Wr_w   = (const float*)d_in[10];
  const float* Wr_b   = (const float*)d_in[11];
  const float* Wlin_w = (const float*)d_in[12];
  const float* Wlin_b = (const float*)d_in[13];
  float* out = (float*)d_out;

  char* ws = (char*)d_ws;
  size_t off = 0;
  auto alloc = [&](size_t bytes) -> char* {
    char* p = ws + off; off += (bytes + 255) & ~(size_t)255; return p;
  };
  __bf16* Wihbf = (__bf16*)alloc((size_t)G_SZ * I_SZ * 2);
  __bf16* Whhbf = (__bf16*)alloc((size_t)G_SZ * H_SZ * 2);
  __bf16* Xbf   = (__bf16*)alloc((size_t)L_SEQ * B_SZ * I_SZ * 2);
  float*  Gx    = (float*) alloc((size_t)L_SEQ * B_SZ * G_SZ * 4);
  float*  Gpart = (float*) alloc((size_t)KSPLIT * B_SZ * G_SZ * 4);
  __bf16* Hbf0  = (__bf16*)alloc((size_t)B_SZ * H_SZ * 2);
  __bf16* Hbf1  = (__bf16*)alloc((size_t)B_SZ * H_SZ * 2);
  float*  Cst   = (float*) alloc((size_t)B_SZ * H_SZ * 4);
  float*  Fm    = (float*) alloc((size_t)B_SZ * SR_SZ * T_SZ * 4);
  float*  qsbuf = (float*) alloc((size_t)L_SEQ * B_SZ * T_SZ * 4);

  int n;
  n = G_SZ * I_SZ;          k_f32_to_bf16<<<(n + 255) / 256, 256, 0, stream>>>(W_ih, Wihbf, n);
  n = G_SZ * H_SZ;          k_f32_to_bf16<<<(n + 255) / 256, 256, 0, stream>>>(W_hh, Whhbf, n);
  n = L_SEQ * B_SZ * I_SZ;  k_f32_to_bf16<<<(n + 255) / 256, 256, 0, stream>>>(inputs, Xbf, n);
  n = B_SZ * H_SZ;          k_f32_to_bf16<<<(n + 255) / 256, 256, 0, stream>>>(h0, Hbf0, n);
  n = B_SZ * H_SZ;          k_copy_f32<<<(n + 255) / 256, 256, 0, stream>>>(c0, Cst, n);
  n = B_SZ * SR_SZ * T_SZ;  k_copy_f32<<<(n + 255) / 256, 256, 0, stream>>>(F0, Fm, n);

  // 128 M-groups * 256 N-groups = 32768 waves, 4 waves/block
  k_gx_gemm<<<8192, 128, 0, stream>>>(Xbf, Wihbf, b_ih, b_hh, Gx);

  for (int t = 0; t < L_SEQ; ++t) {
    const __bf16* hin  = (t & 1) ? Hbf1 : Hbf0;
    __bf16*       hout = (t & 1) ? Hbf0 : Hbf1;
    // 128 col-tiles * 8 K-splits = 1024 waves
    k_lstm_gemm<<<256, 128, 0, stream>>>(hin, Whhbf, Gpart);
    k_lstm_gate<<<128, 256, 0, stream>>>(Gx + (size_t)t * B_SZ * G_SZ, Gpart, Cst, hout,
                                         out + (size_t)t * B_SZ * H_SZ);
    k_fwm_step<<<16, 256, 0, stream>>>(out + (size_t)t * B_SZ * H_SZ, Ww_w, Ww_b, Wr_w, Wr_b,
                                       Fm, qsbuf + (size_t)t * B_SZ * T_SZ);
  }
  n = L_SEQ * B_SZ * H_SZ;
  k_final<<<4096, 256, 0, stream>>>(qsbuf, Wlin_w, Wlin_b, out, n);
}
